// Attention_69123203662341
// MI455X (gfx1250) — matrix operand order repaired and checked by reading, not compiled, not run on hardware
//
#include <hip/hip_runtime.h>
#include <hip/hip_bf16.h>

#define BATCH 32
#define NWIN  513
#define NP    544          // padded N (17 * 32)
#define CDIM  768
#define HEADS 12
#define DH    64
#define CQKV  2304

typedef __bf16 bf16_t;
typedef __attribute__((ext_vector_type(16))) __bf16 v16bf;
typedef __attribute__((ext_vector_type(8)))  __bf16 v8bf;
typedef __attribute__((ext_vector_type(8)))  float  v8f;

union FragU { v16bf f; v8bf h[2]; };

__device__ __forceinline__ v8f wmma_bf16(v16bf a, v16bf b, v8f c) {
  return __builtin_amdgcn_wmma_f32_16x16x32_bf16(false, a, false, b, (short)0, c,
                                                 false, false);
}

// A fragment: 16x32 (MxK) from row-major [M][ld], lane L: M=L%16,
// koff = (L>=16)?8:0, v0-3 = K koff..koff+7, v4-7 = K koff+16..koff+23.
__device__ __forceinline__ v16bf load_frag_a(const bf16_t* base, int ld) {
  int lane = threadIdx.x & 31;
  const bf16_t* p = base + (lane & 15) * ld + ((lane >> 4) << 3);
  FragU u;
  u.h[0] = *(const v8bf*)(p);
  u.h[1] = *(const v8bf*)(p + 16);
  return u.f;
}

// B fragment: 32x16 (KxN) from N-major source [N][ld(K)], lane L:
// N=L%16, koff = (L>=16)?8:0, v0-3 = K koff..koff+7, v4-7 = K koff+16..koff+23.
__device__ __forceinline__ v16bf load_frag_b(const bf16_t* base, int ld) {
  int lane = threadIdx.x & 31;
  const bf16_t* p = base + (lane & 15) * ld + ((lane >> 4) << 3);
  FragU u;
  u.h[0] = *(const v8bf*)(p);
  u.h[1] = *(const v8bf*)(p + 16);
  return u.f;
}

__device__ __forceinline__ float rowred_max(float x) {
  x = fmaxf(x, __shfl_xor(x, 1));
  x = fmaxf(x, __shfl_xor(x, 2));
  x = fmaxf(x, __shfl_xor(x, 4));
  x = fmaxf(x, __shfl_xor(x, 8));
  return x;
}
__device__ __forceinline__ float rowred_sum(float x) {
  x += __shfl_xor(x, 1);
  x += __shfl_xor(x, 2);
  x += __shfl_xor(x, 4);
  x += __shfl_xor(x, 8);
  return x;
}

// ---------------- pack / prep kernels ----------------

__global__ void pack_x_kernel(const float* __restrict__ x, bf16_t* __restrict__ xb) {
  int t = blockIdx.x * blockDim.x + threadIdx.x;
  const int total = BATCH * NP * CDIM;
  if (t >= total) return;
  int c = t % CDIM;
  int r = (t / CDIM) % NP;
  int b = t / (CDIM * NP);
  float v = (r < NWIN) ? x[(b * NWIN + r) * CDIM + c] : 0.0f;
  xb[t] = (bf16_t)v;
}

// dst[c][r] = src[r][c], cast to bf16
__global__ void transpose_pack_kernel(const float* __restrict__ src,
                                      bf16_t* __restrict__ dst,
                                      int rows, int cols) {
  int t = blockIdx.x * blockDim.x + threadIdx.x;
  if (t >= rows * cols) return;
  int r = t / cols;
  int c = t % cols;
  dst[c * rows + r] = (bf16_t)src[t];
}

// bias[h][i][j] (padded to NP x NP, zeros outside NWIN)
__global__ void bias_kernel(const float* __restrict__ table,
                            const int* __restrict__ idx,
                            float* __restrict__ bias) {
  int t = blockIdx.x * blockDim.x + threadIdx.x;
  if (t >= NP * NP) return;
  int j = t % NP;
  int i = t / NP;
  float v[HEADS];
  if (i < NWIN && j < NWIN) {
    int id = idx[i * NWIN + j];
#pragma unroll
    for (int h = 0; h < HEADS; h++) v[h] = table[id * HEADS + h];
  } else {
#pragma unroll
    for (int h = 0; h < HEADS; h++) v[h] = 0.0f;
  }
#pragma unroll
  for (int h = 0; h < HEADS; h++) bias[(h * NP + i) * NP + j] = v[h];
}

// ---------------- GEMM 1: qkv = xb @ wqkv ----------------
// 32x32 tile per wave (2 A-frags x 2 B-frags -> 4 accumulators).
// Writes Q (scaled) and K as [b][h][n][dh] bf16, V transposed [b][h][dh][n].

__global__ void __launch_bounds__(256)
qkv_gemm_kernel(const bf16_t* __restrict__ xb, const bf16_t* __restrict__ wt,
                bf16_t* __restrict__ qb, bf16_t* __restrict__ kb,
                bf16_t* __restrict__ vt) {
  const int CT = CQKV / 32;  // 72
  const int RT = NP / 32;    // 17
  int wv = threadIdx.x >> 5;
  int lane = threadIdx.x & 31;
  int tile = blockIdx.x * 8 + wv;
  int ct = tile % CT;
  int rt = (tile / CT) % RT;
  int b = tile / (CT * RT);

  const bf16_t* abase = xb + (b * NP + rt * 32) * CDIM;
  const bf16_t* bbase = wt + (ct * 32) * CDIM;
  v8f acc[2][2] = {};
  for (int kk = 0; kk < CDIM; kk += 32) {
    v16bf a0 = load_frag_a(abase + kk, CDIM);
    v16bf a1 = load_frag_a(abase + 16 * CDIM + kk, CDIM);
    v16bf b0 = load_frag_b(bbase + kk, CDIM);
    v16bf b1 = load_frag_b(bbase + 16 * CDIM + kk, CDIM);
    acc[0][0] = wmma_bf16(a0, b0, acc[0][0]);
    acc[0][1] = wmma_bf16(a0, b1, acc[0][1]);
    acc[1][0] = wmma_bf16(a1, b0, acc[1][0]);
    acc[1][1] = wmma_bf16(a1, b1, acc[1][1]);
  }
  int col = lane & 15;
  int rowbase = (lane >> 4) * 8;
#pragma unroll
  for (int mi = 0; mi < 2; mi++) {
#pragma unroll
    for (int ni = 0; ni < 2; ni++) {
      v8f a = acc[mi][ni];
      int n0 = ct * 32 + ni * 16;
      int which = n0 / CDIM;
      int local = n0 % CDIM;
      int h = local / DH;
      int d0 = local % DH;
      int nrow = rt * 32 + mi * 16 + rowbase;
      if (which == 0) {
        bf16_t* dst = qb + ((b * HEADS + h) * NP + nrow) * DH + d0 + col;
#pragma unroll
        for (int v = 0; v < 8; v++) dst[v * DH] = (bf16_t)(a[v] * 0.125f);  // Dh^-0.5
      } else if (which == 1) {
        bf16_t* dst = kb + ((b * HEADS + h) * NP + nrow) * DH + d0 + col;
#pragma unroll
        for (int v = 0; v < 8; v++) dst[v * DH] = (bf16_t)a[v];
      } else {
        bf16_t* dst = vt + ((b * HEADS + h) * DH + d0 + col) * NP + nrow;
#pragma unroll
        for (int v = 0; v < 8; v++) dst[v] = (bf16_t)a[v];
      }
    }
  }
}

// ---------------- fused flash attention ----------------
// One wave per (b, h, 16-query tile); streams keys in blocks of 32.

__global__ void __launch_bounds__(256)
attn_kernel(const bf16_t* __restrict__ qb, const bf16_t* __restrict__ kb,
            const bf16_t* __restrict__ vt, const float* __restrict__ bias,
            bf16_t* __restrict__ aout) {
  __shared__ bf16_t lds[8][16 * 40];  // per-wave 16x32 P tile, padded stride 40
  const int QT = NP / 16;  // 34
  int wv = threadIdx.x >> 5;
  int lane = threadIdx.x & 31;
  int tile = blockIdx.x * 8 + wv;
  int qt = tile % QT;
  int h = (tile / QT) % HEADS;
  int b = tile / (QT * HEADS);
  int bh = b * HEADS + h;

  const bf16_t* qbase = qb + (bh * NP + qt * 16) * DH;
  v16bf qa0 = load_frag_a(qbase, DH);
  v16bf qa1 = load_frag_a(qbase + 32, DH);

  int col = lane & 15;
  int rowbase = (lane >> 4) * 8;
  const float* bptr = bias + (h * NP + qt * 16 + rowbase) * NP;

  float m[8], l[8];
  v8f o0 = {}, o1 = {}, o2 = {}, o3 = {};
#pragma unroll
  for (int v = 0; v < 8; v++) { m[v] = -__builtin_inff(); l[v] = 0.0f; }

  for (int jb = 0; jb < NP; jb += 32) {
    const bf16_t* kb0 = kb + (bh * NP + jb) * DH;
    v8f s0 = {}, s1 = {};
    s0 = wmma_bf16(qa0, load_frag_b(kb0, DH), s0);
    s0 = wmma_bf16(qa1, load_frag_b(kb0 + 32, DH), s0);
    const bf16_t* kb1 = kb0 + 16 * DH;
    s1 = wmma_bf16(qa0, load_frag_b(kb1, DH), s1);
    s1 = wmma_bf16(qa1, load_frag_b(kb1 + 32, DH), s1);

    int j0 = jb + col;
    int j1 = jb + 16 + col;
    float p0[8], p1[8];
#pragma unroll
    for (int v = 0; v < 8; v++) {
      float a = (j0 < NWIN) ? s0[v] + bptr[v * NP + j0] : -__builtin_inff();
      float c2 = (j1 < NWIN) ? s1[v] + bptr[v * NP + j1] : -__builtin_inff();
      float mnew = fmaxf(m[v], rowred_max(fmaxf(a, c2)));
      float sc = __expf(m[v] - mnew);
      m[v] = mnew;
      float e0 = __expf(a - mnew);
      float e1 = __expf(c2 - mnew);
      l[v] = l[v] * sc + rowred_sum(e0 + e1);
      o0[v] *= sc; o1[v] *= sc; o2[v] *= sc; o3[v] *= sc;
      p0[v] = e0; p1[v] = e1;
    }
    // stage P (C/D layout) through LDS, reload as A fragment
#pragma unroll
    for (int v = 0; v < 8; v++) {
      lds[wv][(rowbase + v) * 40 + col] = (bf16_t)p0[v];
      lds[wv][(rowbase + v) * 40 + 16 + col] = (bf16_t)p1[v];
    }
    const bf16_t* lp = &lds[wv][(lane & 15) * 40 + ((lane >> 4) << 3)];
    FragU pu;
    pu.h[0] = *(const v8bf*)(lp);
    pu.h[1] = *(const v8bf*)(lp + 16);
    v16bf pa = pu.f;

    const bf16_t* vb = vt + bh * DH * NP + jb;
    o0 = wmma_bf16(pa, load_frag_b(vb + 0 * 16 * NP, NP), o0);
    o1 = wmma_bf16(pa, load_frag_b(vb + 1 * 16 * NP, NP), o1);
    o2 = wmma_bf16(pa, load_frag_b(vb + 2 * 16 * NP, NP), o2);
    o3 = wmma_bf16(pa, load_frag_b(vb + 3 * 16 * NP, NP), o3);
  }

  bf16_t* dst = aout + (b * NP + qt * 16 + rowbase) * CDIM + h * DH;
#pragma unroll
  for (int v = 0; v < 8; v++) {
    float rl = 1.0f / l[v];
    dst[v * CDIM + col] = (bf16_t)(o0[v] * rl);
    dst[v * CDIM + 16 + col] = (bf16_t)(o1[v] * rl);
    dst[v * CDIM + 32 + col] = (bf16_t)(o2[v] * rl);
    dst[v * CDIM + 48 + col] = (bf16_t)(o3[v] * rl);
  }
}

// ---------------- GEMM 3: out = aout @ wproj + b_proj ----------------
// 32x32 tile per wave.

__global__ void __launch_bounds__(256)
proj_gemm_kernel(const bf16_t* __restrict__ aout, const bf16_t* __restrict__ wpt,
                 const float* __restrict__ bproj, float* __restrict__ out) {
  const int CT = CDIM / 32;  // 24
  const int RT = NP / 32;    // 17
  int wv = threadIdx.x >> 5;
  int lane = threadIdx.x & 31;
  int tile = blockIdx.x * 8 + wv;
  int ct = tile % CT;
  int rt = (tile / CT) % RT;
  int b = tile / (CT * RT);

  const bf16_t* abase = aout + (b * NP + rt * 32) * CDIM;
  const bf16_t* bbase = wpt + (ct * 32) * CDIM;
  v8f acc[2][2] = {};
  for (int kk = 0; kk < CDIM; kk += 32) {
    v16bf a0 = load_frag_a(abase + kk, CDIM);
    v16bf a1 = load_frag_a(abase + 16 * CDIM + kk, CDIM);
    v16bf b0 = load_frag_b(bbase + kk, CDIM);
    v16bf b1 = load_frag_b(bbase + 16 * CDIM + kk, CDIM);
    acc[0][0] = wmma_bf16(a0, b0, acc[0][0]);
    acc[0][1] = wmma_bf16(a0, b1, acc[0][1]);
    acc[1][0] = wmma_bf16(a1, b0, acc[1][0]);
    acc[1][1] = wmma_bf16(a1, b1, acc[1][1]);
  }
  int col = lane & 15;
  int rowbase = (lane >> 4) * 8;
#pragma unroll
  for (int mi = 0; mi < 2; mi++) {
#pragma unroll
    for (int ni = 0; ni < 2; ni++) {
      v8f a = acc[mi][ni];
      int n0 = ct * 32 + ni * 16;
      float bv = bproj[n0 + col];
#pragma unroll
      for (int v = 0; v < 8; v++) {
        int r = rt * 32 + mi * 16 + rowbase + v;
        if (r < NWIN) out[(b * NWIN + r) * CDIM + n0 + col] = a[v] + bv;
      }
    }
  }
}

// ---------------- host launch ----------------

static constexpr size_t ALN(size_t x) { return (x + 255) & ~(size_t)255; }
static constexpr size_t SZ_XB = (size_t)BATCH * NP * CDIM * 2;           // also aout
static constexpr size_t SZ_WQT = (size_t)CQKV * CDIM * 2;
static constexpr size_t SZ_WPT = (size_t)CDIM * CDIM * 2;
static constexpr size_t SZ_HB = (size_t)BATCH * HEADS * NP * DH * 2;
static constexpr size_t O_XB = 0;
static constexpr size_t O_WQT = O_XB + ALN(SZ_XB);
static constexpr size_t O_WPT = O_WQT + ALN(SZ_WQT);
static constexpr size_t O_Q = O_WPT + ALN(SZ_WPT);
static constexpr size_t O_K = O_Q + ALN(SZ_HB);
static constexpr size_t O_VT = O_K + ALN(SZ_HB);
static constexpr size_t O_BIAS = O_VT + ALN(SZ_HB);

extern "C" void kernel_launch(void* const* d_in, const int* in_sizes, int n_in,
                              void* d_out, int out_size, void* d_ws, size_t ws_size,
                              hipStream_t stream) {
  const float* x = (const float*)d_in[0];
  const float* w_qkv = (const float*)d_in[1];
  const float* w_proj = (const float*)d_in[2];
  const float* b_proj = (const float*)d_in[3];
  const float* rel_tab = (const float*)d_in[4];
  const int* rel_idx = (const int*)d_in[5];
  char* ws = (char*)d_ws;

  bf16_t* xb = (bf16_t*)(ws + O_XB);      // reused as attention output (aout)
  bf16_t* wqt = (bf16_t*)(ws + O_WQT);
  bf16_t* wpt = (bf16_t*)(ws + O_WPT);
  bf16_t* qb = (bf16_t*)(ws + O_Q);
  bf16_t* kb = (bf16_t*)(ws + O_K);
  bf16_t* vt = (bf16_t*)(ws + O_VT);
  float* bias = (float*)(ws + O_BIAS);
  float* out = (float*)d_out;

  pack_x_kernel<<<(BATCH * NP * CDIM + 255) / 256, 256, 0, stream>>>(x, xb);
  transpose_pack_kernel<<<(CDIM * CQKV + 255) / 256, 256, 0, stream>>>(w_qkv, wqt,
                                                                       CDIM, CQKV);
  transpose_pack_kernel<<<(CDIM * CDIM + 255) / 256, 256, 0, stream>>>(w_proj, wpt,
                                                                       CDIM, CDIM);
  bias_kernel<<<(NP * NP + 255) / 256, 256, 0, stream>>>(rel_tab, rel_idx, bias);

  qkv_gemm_kernel<<<BATCH * (NP / 32) * (CQKV / 32) / 8, 256, 0, stream>>>(
      xb, wqt, qb, kb, vt);
  attn_kernel<<<BATCH * HEADS * (NP / 16) / 8, 256, 0, stream>>>(qb, kb, vt, bias,
                                                                 xb /* aout */);
  proj_gemm_kernel<<<BATCH * (NP / 32) * (CDIM / 32) / 8, 256, 0, stream>>>(
      xb, wpt, b_proj, out);
}
